// MoE_80376017977412
// MI455X (gfx1250) — matrix operand order repaired and checked by reading, not compiled, not run on hardware
//
#include <hip/hip_runtime.h>

// ---- problem constants (from reference) ----
#define T_   2048
#define D_   2048
#define I_   1408
#define SH_  2816
#define E_   64
#define K_   6
#define CAP_ 384
#define NASSIGN (T_ * K_)

// ---- WMMA vector types (gfx1250, wave32) ----
typedef __attribute__((ext_vector_type(16))) __bf16 v16bf;
typedef __attribute__((ext_vector_type(8)))  float  v8f;

union BF16Frag { uint4 q[2]; v16bf v; };
union FragC    { v8f v; float f[8]; };

__device__ __forceinline__ unsigned int f2bf_u(float f) {
  union { float f; unsigned int u; } v; v.f = f;
  unsigned int u = v.u;
  u += 0x7FFFu + ((u >> 16) & 1u);   // round-to-nearest-even
  return u >> 16;
}
__device__ __forceinline__ unsigned short f2bf(float f) { return (unsigned short)f2bf_u(f); }

__device__ __forceinline__ void store8bf(unsigned short* dst, float4 a, float4 b) {
  uint4 o;
  o.x = f2bf_u(a.x) | (f2bf_u(a.y) << 16);
  o.y = f2bf_u(a.z) | (f2bf_u(a.w) << 16);
  o.z = f2bf_u(b.x) | (f2bf_u(b.y) << 16);
  o.w = f2bf_u(b.z) | (f2bf_u(b.w) << 16);
  *(uint4*)dst = o;
}

// silu via v_exp + v_rcp (output is rounded to bf16, 1-ulp rcp is plenty)
__device__ __forceinline__ float silu_fast(float g) {
  return g * __builtin_amdgcn_rcpf(1.0f + __expf(-g));
}

// ---- tiling ----
#define BM 128
#define BN 64
#define BK 32
#define LDA 40   // bf16 elems per LDS row: 32 + 8 pad (80B, 16B aligned)
#define LDB 40

// ============================ dispatch ============================
__global__ void dispatch_init_kernel(int* counts, int* toktab) {
  int i = blockIdx.x * blockDim.x + threadIdx.x;
  if (i < E_ * CAP_) toktab[i] = -1;
  if (i < E_) counts[i] = 0;
}

__global__ void dispatch_build_kernel(const int* __restrict__ idx, int* counts,
                                      int* toktab, int* pos) {
  int a = blockIdx.x * blockDim.x + threadIdx.x;
  if (a >= NASSIGN) return;
  int e = idx[a];
  int t = a / K_;
  int r = atomicAdd(&counts[e], 1);
  if (r < CAP_) { toktab[e * CAP_ + r] = t; pos[a] = e * CAP_ + r; }
  else          { pos[a] = -1; }
}

// ============================ x -> bf16 (+zero pad row) ============================
__global__ void cvt_x_kernel(const float* __restrict__ x, unsigned short* __restrict__ xb) {
  int i = blockIdx.x * blockDim.x + threadIdx.x;
  if (i >= (T_ + 1) * D_) return;
  float v = (i < T_ * D_) ? x[i] : 0.0f;
  xb[i] = f2bf(v);
}

// ============================ fused gate/up GEMM ============================
// h[e, m, n] = silu(Xg @ W1^T) * (Xg @ W3^T);  Xg rows gathered via tab (or identity)
__global__ __launch_bounds__(256)
void moe_gateup_kernel(const unsigned short* __restrict__ xb,  // [(T+1), Kd] bf16
                       const int* __restrict__ tab,            // [E, Mrows] or null
                       const float* __restrict__ W1,           // [E, N, Kd] fp32
                       const float* __restrict__ W3,
                       unsigned short* __restrict__ hout,      // [E, Mrows, N] bf16
                       int Mrows, int N, int Kd, int zeroRow) {
  int e   = blockIdx.z;
  int m0b = blockIdx.y * BM;
  int n0b = blockIdx.x * BN;
  const float* W1e = W1 + (size_t)e * N * Kd;
  const float* W3e = W3 + (size_t)e * N * Kd;
  unsigned short* houte = hout + (size_t)e * Mrows * N;

  __shared__ unsigned short As[BM * LDA];
  __shared__ unsigned short Bs1[BN * LDB];
  __shared__ unsigned short Bs3[BN * LDB];

  int tid = threadIdx.x;
  int lane = tid & 31, wid = tid >> 5;
  int waveM = wid & 3, waveN = wid >> 2;        // 4 x 2 wave grid
  int half = lane >> 4, l16 = lane & 15;

  // A loader: each thread copies 16 bf16 of one row
  int arow = tid >> 1, aseg = tid & 1;
  int srcRow = m0b + arow;
  if (tab) { int tt = tab[(size_t)e * Mrows + srcRow]; srcRow = (tt < 0) ? zeroRow : tt; }
  const unsigned short* aSrc = xb + (size_t)srcRow * Kd + aseg * 16;

  // B loader: each thread converts 8 floats of one row
  int brow = tid >> 2, bseg = tid & 3;
  const float* b1Src = W1e + (size_t)(n0b + brow) * Kd + bseg * 8;
  const float* b3Src = W3e + (size_t)(n0b + brow) * Kd + bseg * 8;

  FragC acc1[2][2], acc3[2][2];
  for (int i = 0; i < 2; i++)
    for (int j = 0; j < 2; j++)
      for (int r = 0; r < 8; r++) { acc1[i][j].f[r] = 0.f; acc3[i][j].f[r] = 0.f; }

  for (int k0 = 0; k0 < Kd; k0 += BK) {
    // prefetch weight stream two K-tiles ahead (global_prefetch_b8)
    if (k0 + 2 * BK < Kd) {
      __builtin_prefetch(b1Src + k0 + 2 * BK, 0, 1);
      __builtin_prefetch(b3Src + k0 + 2 * BK, 0, 1);
      __builtin_prefetch(aSrc + k0 + 2 * BK, 0, 1);
    }
    { // stage A (bf16 copy)
      const uint4* s = (const uint4*)(aSrc + k0);
      uint4* d = (uint4*)(&As[arow * LDA + aseg * 16]);
      d[0] = s[0]; d[1] = s[1];
    }
    { // stage B1/B3 (fp32 -> bf16 on the fly: weights read once from HBM)
      float4 f0 = *(const float4*)(b1Src + k0);
      float4 f1 = *(const float4*)(b1Src + k0 + 4);
      store8bf(&Bs1[brow * LDB + bseg * 8], f0, f1);
      f0 = *(const float4*)(b3Src + k0);
      f1 = *(const float4*)(b3Src + k0 + 4);
      store8bf(&Bs3[brow * LDB + bseg * 8], f0, f1);
    }
    __syncthreads();

    BF16Frag a[2], b1[2], b3[2];
    for (int i = 0; i < 2; i++) {   // A 16x32 layout: lanes0-15 K0..7,16..23; lanes16-31 +8
      const unsigned short* rp = &As[(waveM * 32 + i * 16 + l16) * LDA];
      a[i].q[0] = *(const uint4*)(rp + 8 * half);
      a[i].q[1] = *(const uint4*)(rp + 16 + 8 * half);
    }
    for (int j = 0; j < 2; j++) {   // B 32x16 layout: lanes0-15 K0..7,16..23; lanes16-31 +8
      const unsigned short* r1 = &Bs1[(waveN * 32 + j * 16 + l16) * LDB];
      b1[j].q[0] = *(const uint4*)(r1 + 8 * half);
      b1[j].q[1] = *(const uint4*)(r1 + 16 + 8 * half);
      const unsigned short* r3 = &Bs3[(waveN * 32 + j * 16 + l16) * LDB];
      b3[j].q[0] = *(const uint4*)(r3 + 8 * half);
      b3[j].q[1] = *(const uint4*)(r3 + 16 + 8 * half);
    }
    for (int i = 0; i < 2; i++)
      for (int j = 0; j < 2; j++) {
        acc1[i][j].v = __builtin_amdgcn_wmma_f32_16x16x32_bf16(
            false, a[i].v, false, b1[j].v, (short)0, acc1[i][j].v, false, false);
        acc3[i][j].v = __builtin_amdgcn_wmma_f32_16x16x32_bf16(
            false, a[i].v, false, b3[j].v, (short)0, acc3[i][j].v, false, false);
      }
    __syncthreads();
  }

  // epilogue: h = silu(g) * u, bf16 out.  C layout: elem r -> M = r + 8*half, N = lane&15
  for (int i = 0; i < 2; i++)
    for (int j = 0; j < 2; j++)
      for (int r = 0; r < 8; r++) {
        int m = m0b + waveM * 32 + i * 16 + r + 8 * half;
        int n = n0b + waveN * 32 + j * 16 + l16;
        float h = silu_fast(acc1[i][j].f[r]) * acc3[i][j].f[r];
        houte[(size_t)m * N + n] = f2bf(h);
      }
}

// ============================ down GEMM ============================
// y[e, m, n] = H[e, m, :] @ W2[e, n, :]^T    (fp32 out)
__global__ __launch_bounds__(256)
void moe_down_kernel(const unsigned short* __restrict__ hin,   // [E, Mrows, Kd] bf16
                     const float* __restrict__ W2,             // [E, N, Kd] fp32
                     float* __restrict__ yout,                 // [E, Mrows, N] fp32
                     int Mrows, int N, int Kd) {
  int e   = blockIdx.z;
  int m0b = blockIdx.y * BM;
  int n0b = blockIdx.x * BN;
  const unsigned short* hine = hin + (size_t)e * Mrows * Kd;
  const float* W2e = W2 + (size_t)e * N * Kd;
  float* youte = yout + (size_t)e * Mrows * N;

  __shared__ unsigned short As[BM * LDA];
  __shared__ unsigned short Bs[BN * LDB];

  int tid = threadIdx.x;
  int lane = tid & 31, wid = tid >> 5;
  int waveM = wid & 3, waveN = wid >> 2;
  int half = lane >> 4, l16 = lane & 15;

  int arow = tid >> 1, aseg = tid & 1;
  const unsigned short* aSrc = hine + (size_t)(m0b + arow) * Kd + aseg * 16;
  int brow = tid >> 2, bseg = tid & 3;
  const float* bSrc = W2e + (size_t)(n0b + brow) * Kd + bseg * 8;

  FragC acc[2][2];
  for (int i = 0; i < 2; i++)
    for (int j = 0; j < 2; j++)
      for (int r = 0; r < 8; r++) acc[i][j].f[r] = 0.f;

  for (int k0 = 0; k0 < Kd; k0 += BK) {
    if (k0 + 2 * BK < Kd) {
      __builtin_prefetch(bSrc + k0 + 2 * BK, 0, 1);
      __builtin_prefetch(aSrc + k0 + 2 * BK, 0, 1);
    }
    {
      const uint4* s = (const uint4*)(aSrc + k0);
      uint4* d = (uint4*)(&As[arow * LDA + aseg * 16]);
      d[0] = s[0]; d[1] = s[1];
    }
    {
      float4 f0 = *(const float4*)(bSrc + k0);
      float4 f1 = *(const float4*)(bSrc + k0 + 4);
      store8bf(&Bs[brow * LDB + bseg * 8], f0, f1);
    }
    __syncthreads();

    BF16Frag a[2], b[2];
    for (int i = 0; i < 2; i++) {
      const unsigned short* rp = &As[(waveM * 32 + i * 16 + l16) * LDA];
      a[i].q[0] = *(const uint4*)(rp + 8 * half);
      a[i].q[1] = *(const uint4*)(rp + 16 + 8 * half);
    }
    for (int j = 0; j < 2; j++) {
      const unsigned short* rp = &Bs[(waveN * 32 + j * 16 + l16) * LDB];
      b[j].q[0] = *(const uint4*)(rp + 8 * half);
      b[j].q[1] = *(const uint4*)(rp + 16 + 8 * half);
    }
    for (int i = 0; i < 2; i++)
      for (int j = 0; j < 2; j++)
        acc[i][j].v = __builtin_amdgcn_wmma_f32_16x16x32_bf16(
            false, a[i].v, false, b[j].v, (short)0, acc[i][j].v, false, false);
    __syncthreads();
  }

  for (int i = 0; i < 2; i++)
    for (int j = 0; j < 2; j++)
      for (int r = 0; r < 8; r++) {
        int m = m0b + waveM * 32 + i * 16 + r + 8 * half;
        int n = n0b + waveN * 32 + j * 16 + l16;
        youte[(size_t)m * N + n] = acc[i][j].f[r];
      }
}

// ============================ combine ============================
// out[t, :] = z[t, :] + sum_k w[t,k] * yg[pos[t,k], :]   (fixed k order -> deterministic)
__global__ __launch_bounds__(256)
void combine_kernel(const float* __restrict__ zbuf, const float* __restrict__ yg,
                    const float* __restrict__ wts, const int* __restrict__ pos,
                    float* __restrict__ out) {
  int t = blockIdx.x;
  float w[K_]; int p[K_];
  for (int k = 0; k < K_; k++) { p[k] = pos[t * K_ + k]; w[k] = wts[t * K_ + k]; }
  for (int d4 = threadIdx.x; d4 < D_ / 4; d4 += blockDim.x) {
    float4 acc = *(const float4*)(zbuf + (size_t)t * D_ + 4 * d4);
    for (int k = 0; k < K_; k++) {
      if (p[k] >= 0) {
        float4 v = *(const float4*)(yg + (size_t)p[k] * D_ + 4 * d4);
        acc.x += w[k] * v.x; acc.y += w[k] * v.y;
        acc.z += w[k] * v.z; acc.w += w[k] * v.w;
      }
    }
    *(float4*)(out + (size_t)t * D_ + 4 * d4) = acc;
  }
}

// ============================ host launcher ============================
extern "C" void kernel_launch(void* const* d_in, const int* in_sizes, int n_in,
                              void* d_out, int out_size, void* d_ws, size_t ws_size,
                              hipStream_t stream) {
  (void)in_sizes; (void)n_in; (void)out_size; (void)ws_size;
  const float* x   = (const float*)d_in[0];
  const float* wts = (const float*)d_in[1];
  const int*   idx = (const int*)d_in[2];
  const float* W1  = (const float*)d_in[3];
  const float* W2  = (const float*)d_in[4];
  const float* W3  = (const float*)d_in[5];
  const float* sw1 = (const float*)d_in[6];
  const float* sw2 = (const float*)d_in[7];
  const float* sw3 = (const float*)d_in[8];
  float* out = (float*)d_out;

  char* ws = (char*)d_ws;
  size_t off = 0;
  unsigned short* xb = (unsigned short*)(ws + off); off += (size_t)(T_ + 1) * D_ * 2;
  int* counts = (int*)(ws + off); off += 256;
  int* toktab = (int*)(ws + off); off += (size_t)E_ * CAP_ * 4;
  int* pos    = (int*)(ws + off); off += (size_t)NASSIGN * 4;
  unsigned short* hbuf  = (unsigned short*)(ws + off); off += (size_t)E_ * CAP_ * I_ * 2;
  unsigned short* hsbuf = (unsigned short*)(ws + off); off += (size_t)T_ * SH_ * 2;
  float* yg   = (float*)(ws + off); off += (size_t)E_ * CAP_ * D_ * 4;
  float* zbuf = (float*)(ws + off); off += (size_t)T_ * D_ * 4;

  dispatch_init_kernel<<<(E_ * CAP_ + 255) / 256, 256, 0, stream>>>(counts, toktab);
  dispatch_build_kernel<<<(NASSIGN + 255) / 256, 256, 0, stream>>>(idx, counts, toktab, pos);
  cvt_x_kernel<<<((T_ + 1) * D_ + 255) / 256, 256, 0, stream>>>(x, xb);

  // routed gate/up:   [E=64] x (CAP x I) = Xg(CAP x D) @ W1/W3(I x D)^T
  moe_gateup_kernel<<<dim3(I_ / BN, CAP_ / BM, E_), 256, 0, stream>>>(
      xb, toktab, W1, W3, hbuf, CAP_, I_, D_, T_);
  // shared gate/up:   (T x SH) = X(T x D) @ sw1/sw3(SH x D)^T
  moe_gateup_kernel<<<dim3(SH_ / BN, T_ / BM, 1), 256, 0, stream>>>(
      xb, nullptr, sw1, sw3, hsbuf, T_, SH_, D_, T_);
  // routed down:      [E] x (CAP x D) = H(CAP x I) @ W2(D x I)^T
  moe_down_kernel<<<dim3(D_ / BN, CAP_ / BM, E_), 256, 0, stream>>>(
      hbuf, W2, yg, CAP_, D_, I_);
  // shared down:      (T x D) = Hs(T x SH) @ sw2(D x SH)^T
  moe_down_kernel<<<dim3(D_ / BN, T_ / BM, 1), 256, 0, stream>>>(
      hsbuf, sw2, zbuf, T_, D_, SH_);
  // weighted combine + shared add
  combine_kernel<<<T_, 256, 0, stream>>>(zbuf, yg, wts, pos, out);
}
